// LocalSimGuideSampler_55791625175245
// MI455X (gfx1250) — matrix operand order repaired and checked by reading, not compiled, not run on hardware
//
#include <hip/hip_runtime.h>

// ---------------------------------------------------------------------------
// LocalSimGuideSampler for MI455X (gfx1250, wave32).
// Bandwidth-bound pipeline (~210MB traffic, <1 GFLOP) -> f32 data end-to-end,
// WMMA f16->f32 only for the tiny 1x1-conv GEMMs (K=72 padded to 96).
// ---------------------------------------------------------------------------

typedef _Float16 v16h __attribute__((ext_vector_type(16)));
typedef float    v8f  __attribute__((ext_vector_type(8)));

#define EPS_GN  1e-5f
#define EPS_COS 1e-8f

constexpr int  Bn = 2, Cc = 64, Hl = 128, Wl = 128, Hh2 = 256, Wh2 = 256;
constexpr int  KPAD  = 96;    // CIN=72 padded to 3 x K32 WMMA steps
constexpr int  SIMLD = 104;   // LDS row stride (halfs), even, some conflict pad
constexpr int  TP    = 128;   // pixels per block
constexpr long NL    = (long)Bn * 32 * Hl * Wl;   // per-buffer elems (B,32,128,128)

// ---------------------------------------------------------------------------
// Kernel 1: GroupNorm statistics (mean, rstd) per (batch, group-of-8-channels)
// blocks 0..15 -> x_l, 16..31 -> x_h
// stats layout: [mean_l 16][rstd_l 16][mean_h 16][rstd_h 16]
// ---------------------------------------------------------------------------
__global__ __launch_bounds__(256)
void gn_stats_kernel(const float* __restrict__ xl, const float* __restrict__ xh,
                     float* __restrict__ stats) {
  const int  j    = blockIdx.x;
  const bool high = j >= 16;
  const int  bg   = j & 15;
  const int  b    = bg >> 3, g = bg & 7;
  const float* x  = high ? xh : xl;
  const long HW   = high ? 65536 : 16384;
  const long base = ((long)b * 64 + g * 8) * HW;
  const long n    = 8L * HW;

  float s = 0.f, ss = 0.f;
  for (long i = threadIdx.x; i < n; i += 256) {
    float v = x[base + i];
    s += v; ss += v * v;
  }
  __shared__ float rs[256], rss[256];
  rs[threadIdx.x] = s; rss[threadIdx.x] = ss;
  __syncthreads();
  for (int st = 128; st > 0; st >>= 1) {
    if (threadIdx.x < st) {
      rs[threadIdx.x]  += rs[threadIdx.x + st];
      rss[threadIdx.x] += rss[threadIdx.x + st];
    }
    __syncthreads();
  }
  if (threadIdx.x == 0) {
    float mean = rs[0] / (float)n;
    float var  = rss[0] / (float)n - mean * mean;
    stats[(high ? 32 : 0) + bg]      = mean;
    stats[(high ? 32 : 0) + 16 + bg] = rsqrtf(var + EPS_GN);
  }
}

// ---------------------------------------------------------------------------
// Kernel 2: fused GN-apply + 8-neighbor cosine similarity + dual 1x1 conv
// (W_off stacked over W_ds -> M rows) via v_wmma_f32_16x16x32_f16.
// One block = 128 pixels of one row; 128 threads = 4 waves.
// HIGH results are scattered through pixel_unshuffle into (B,32,128,128).
// ---------------------------------------------------------------------------
template <bool HIGH>
__global__ __launch_bounds__(128)
void sim_conv_kernel(const float* __restrict__ x,
                     const float* __restrict__ gnw, const float* __restrict__ gnb,
                     const float* __restrict__ stats,
                     const float* __restrict__ w_off, const float* __restrict__ s_off,
                     const float* __restrict__ b_off,
                     const float* __restrict__ w_ds,  const float* __restrict__ s_ds,
                     const float* __restrict__ b_ds,
                     float* __restrict__ offOut, float* __restrict__ dsOut) {
  constexpr int Hd   = HIGH ? 256 : 128;
  constexpr int Wd   = HIGH ? 256 : 128;
  constexpr int M    = HIGH ? 16 : 64;   // off rows + ds rows
  constexpr int MO   = M / 2;
  constexpr int SEGS = Wd / TP;

  __shared__ __align__(16) _Float16 simS[TP][SIMLD];  // per-pixel 96-ch features
  __shared__ __align__(16) _Float16 wS[64][SIMLD];    // stacked weights (f16)
  __shared__ float scS[64], biS[64];                  // post-GEMM scale/bias
  __shared__ float chScale[64], chShift[64];          // GN fused affine

  const int tid = threadIdx.x;
  const int seg = blockIdx.x % SEGS;
  const int y   = (blockIdx.x / SEGS) % Hd;
  const int b   =  blockIdx.x / (SEGS * Hd);

  if (tid < 64) {
    int   g    = tid >> 3;
    float mean = stats[b * 8 + g];
    float rstd = stats[16 + b * 8 + g];
    float sc   = rstd * gnw[tid];
    chScale[tid] = sc;
    chShift[tid] = gnb[tid] - mean * sc;
  }
  for (int idx = tid; idx < M * KPAD; idx += 128) {
    int m = idx / KPAD, k = idx % KPAD;
    float wv = 0.f;
    if (k < 72) wv = (m < MO) ? w_off[m * 72 + k] : w_ds[(m - MO) * 72 + k];
    wS[m][k] = (_Float16)wv;
  }
  if (tid < M) {
    scS[tid] = (tid < MO) ? s_off[tid] : s_ds[tid - MO];
    biS[tid] = (tid < MO) ? b_off[tid] : b_ds[tid - MO];
  }
  __syncthreads();

  // ---- Phase 1: one pixel per thread -> sim features into LDS (f16) ----
  {
    const int    p  = tid;
    const int    xx = seg * TP + p;
    const float* xb = x + (long)b * Cc * (long)(Hd * Wd);

    float xc[64];
    float cn2 = 0.f;
#pragma unroll
    for (int c = 0; c < 64; ++c) {
      float v = xb[(long)c * (Hd * Wd) + (long)y * Wd + xx] * chScale[c] + chShift[c];
      xc[c] = v;
      cn2  += v * v;
      simS[p][c] = (_Float16)v;
    }
    float cn = sqrtf(cn2);

    int si = 0;
#pragma unroll
    for (int dy = -2; dy <= 2; dy += 2) {
#pragma unroll
      for (int dx = -2; dx <= 2; dx += 2) {
        if (dy == 0 && dx == 0) continue;
        int   ny = y + dy, nx = xx + dx;
        float simv = 0.f;
        if (ny >= 0 && ny < Hd && nx >= 0 && nx < Wd) {
          float dot = 0.f, nn2 = 0.f;
#pragma unroll
          for (int c = 0; c < 64; ++c) {
            float nv = xb[(long)c * (Hd * Wd) + (long)ny * Wd + nx] * chScale[c] + chShift[c];
            dot += xc[c] * nv;
            nn2 += nv * nv;
          }
          simv = dot / fmaxf(cn * sqrtf(nn2), EPS_COS);
        }
        simS[p][64 + si] = (_Float16)simv;
        ++si;
      }
    }
#pragma unroll
    for (int k = 72; k < KPAD; ++k) simS[p][k] = (_Float16)0.f;
  }
  __syncthreads();

  // ---- Phase 2: GEMM D[M x 128] = W[M x 96] * Sim[96 x 128] via WMMA ----
  const int wvid = tid >> 5;
  const int lane = tid & 31;
  const int hi   = lane >> 4;  // K-half selector (wave32 layout)
  const int ln   = lane & 15;
  constexpr int nM     = M / 16;
  constexpr int nTiles = nM * (TP / 16);   // 32 (low) / 8 (high), both /4

  for (int t = wvid; t < nTiles; t += 4) {
    const int mT = t % nM;
    const int nT = t / nM;
    v8f acc = {};
#pragma unroll
    for (int kT = 0; kT < 3; ++kT) {
      union { v16h v; unsigned int u[8]; } A, Bf;
      const int am = mT * 16 + ln;
      const int bn = nT * 16 + ln;
#pragma unroll
      for (int pe = 0; pe < 8; ++pe) {
        // A (16x32 f16): lane row=M, elem pair pe -> K = hi*8 + 2pe (+8 if pe>=4)
        int ka = kT * 32 + hi * 8 + 2 * pe + ((pe >= 4) ? 8 : 0);
        A.u[pe] = *(const unsigned int*)&wS[am][ka];
        // B (32x16 f16): lane col=N, elem pair pe -> K = hi*8 + 2pe (+8 if pe>=4)
        int kb = kT * 32 + hi * 8 + 2 * pe + ((pe >= 4) ? 8 : 0);
        Bf.u[pe] = *(const unsigned int*)&simS[bn][kb];
      }
      acc = __builtin_amdgcn_wmma_f32_16x16x32_f16(false, A.v, false, Bf.v,
                                                   (short)0, acc, false, false);
    }
    // C/D layout: VGPR r, lanes 0-15 -> M=r, lanes 16-31 -> M=8+r, N=lane%16
#pragma unroll
    for (int r = 0; r < 8; ++r) {
      int   m   = mT * 16 + hi * 8 + r;
      int   n   = nT * 16 + ln;
      int   xx  = seg * TP + n;
      float val = acc[r] * scS[m] + biS[m];
      if (!HIGH) {
        if (m < 32) offOut[(((long)b * 32 + m) * 128 + y) * 128 + xx] = val;
        else        dsOut[(((long)b * 32 + (m - 32)) * 128 + y) * 128 + xx] = val;
      } else {
        int  c  = (m < 8) ? m : (m - 8);
        int  co = c * 4 + (y & 1) * 2 + (xx & 1);   // pixel_unshuffle channel
        long oi = (((long)b * 32 + co) * 128 + (y >> 1)) * 128 + (xx >> 1);
        if (m < 8) offOut[oi] = val; else dsOut[oi] = val;
      }
    }
  }
}

// ---------------------------------------------------------------------------
// Kernel 3: offset = (offL+offHu) * sigmoid(dsL+dsHu) + init_pos
// ---------------------------------------------------------------------------
__global__ __launch_bounds__(256)
void combine_kernel(const float* __restrict__ offL, const float* __restrict__ dsL,
                    const float* __restrict__ offHu, const float* __restrict__ dsHu,
                    float* __restrict__ offset) {
  long idx = (long)blockIdx.x * blockDim.x + threadIdx.x;
  if (idx >= NL) return;
  int   c    = (int)((idx >> 14) & 31);       // (B,32,128,128)
  float off  = offL[idx] + offHu[idx];
  float ds   = dsL[idx] + dsHu[idx];
  float gate = 1.f / (1.f + __expf(-ds));
  int d = c >> 4, q = c & 15;
  int a = (q >> 1) & 1, bb = q & 1;
  float init = (d == 0) ? (bb ? 0.25f : -0.25f) : (a ? 0.25f : -0.25f);
  offset[idx] = off * gate + init;
}

// ---------------------------------------------------------------------------
// Kernel 4: deformable bilinear sample with border clamp.
// One thread per (b, g, Y, X); streams 64 group channels.
// ---------------------------------------------------------------------------
__global__ __launch_bounds__(256)
void sample_kernel(const float* __restrict__ feat, const float* __restrict__ offset,
                   float* __restrict__ out) {
  long idx = (long)blockIdx.x * blockDim.x + threadIdx.x;  // B*G*256*256
  const int X = (int)(idx & 255);
  const int Y = (int)((idx >> 8) & 255);
  const int g = (int)((idx >> 16) & 3);
  const int b = (int)(idx >> 18);

  const int  q     = g * 4 + (Y & 1) * 2 + (X & 1);
  const int  y     = Y >> 1, x = X >> 1;
  const long obase = (((long)b * 32 + q) * 128 + y) * 128 + x;
  float offx = offset[obase];
  float offy = offset[obase + (long)16 * 128 * 128];

  float gx = fminf(fmaxf((float)x + offx, 0.f), 127.f);
  float gy = fminf(fmaxf((float)y + offy, 0.f), 127.f);
  float x0 = floorf(gx), y0 = floorf(gy);
  float wx = gx - x0,    wy = gy - y0;
  int x0i = (int)x0, y0i = (int)y0;
  int x1i = (x0i + 1 > 127) ? 127 : x0i + 1;
  int y1i = (y0i + 1 > 127) ? 127 : y0i + 1;
  float w00 = (1.f - wx) * (1.f - wy), w01 = wx * (1.f - wy);
  float w10 = (1.f - wx) * wy,         w11 = wx * wy;

  const float* fb  = feat + ((long)b * 256 + g * 64) * 16384;
  const float* p00 = fb + y0i * 128 + x0i;
  const float* p01 = fb + y0i * 128 + x1i;
  const float* p10 = fb + y1i * 128 + x0i;
  const float* p11 = fb + y1i * 128 + x1i;
  float* ob = out + (((long)b * 256 + g * 64) * 256 + Y) * 256 + X;

  __builtin_prefetch(p00, 0, 0);
  __builtin_prefetch(p11, 0, 0);
#pragma unroll 8
  for (int ci = 0; ci < 64; ++ci) {
    float v = w00 * p00[(long)ci * 16384] + w01 * p01[(long)ci * 16384] +
              w10 * p10[(long)ci * 16384] + w11 * p11[(long)ci * 16384];
    ob[(long)ci * 65536] = v;
  }
}

// ---------------------------------------------------------------------------
extern "C" void kernel_launch(void* const* d_in, const int* in_sizes, int n_in,
                              void* d_out, int out_size, void* d_ws, size_t ws_size,
                              hipStream_t stream) {
  const float* x_l     = (const float*)d_in[0];
  const float* x_h     = (const float*)d_in[1];
  const float* feat    = (const float*)d_in[2];
  const float* gn_l_w  = (const float*)d_in[3];
  const float* gn_l_b  = (const float*)d_in[4];
  const float* gn_h_w  = (const float*)d_in[5];
  const float* gn_h_b  = (const float*)d_in[6];
  const float* w_off   = (const float*)d_in[7];
  const float* s_off   = (const float*)d_in[8];
  const float* b_off   = (const float*)d_in[9];
  const float* w_off_h = (const float*)d_in[10];
  const float* s_off_h = (const float*)d_in[11];
  const float* b_off_h = (const float*)d_in[12];
  const float* w_ds    = (const float*)d_in[13];
  const float* s_ds    = (const float*)d_in[14];
  const float* b_ds    = (const float*)d_in[15];
  const float* w_ds_h  = (const float*)d_in[16];
  const float* s_ds_h  = (const float*)d_in[17];
  const float* b_ds_h  = (const float*)d_in[18];
  float* out = (float*)d_out;
  float* ws  = (float*)d_ws;

  // workspace layout (floats): stats[64] @0, then 5 buffers of NL each
  float* stats  = ws;
  float* offL   = ws + 256;
  float* dsL    = offL + NL;
  float* offHu  = dsL + NL;
  float* dsHu   = offHu + NL;
  float* offset = dsHu + NL;

  gn_stats_kernel<<<32, 256, 0, stream>>>(x_l, x_h, stats);

  sim_conv_kernel<false><<<Bn * Hl * 1, 128, 0, stream>>>(
      x_l, gn_l_w, gn_l_b, stats,
      w_off, s_off, b_off, w_ds, s_ds, b_ds, offL, dsL);

  sim_conv_kernel<true><<<Bn * Hh2 * 2, 128, 0, stream>>>(
      x_h, gn_h_w, gn_h_b, stats + 32,
      w_off_h, s_off_h, b_off_h, w_ds_h, s_ds_h, b_ds_h, offHu, dsHu);

  combine_kernel<<<(int)((NL + 255) / 256), 256, 0, stream>>>(offL, dsL, offHu, dsHu, offset);

  sample_kernel<<<2048, 256, 0, stream>>>(feat, offset, out);
}
